// PrunedKVAttention_30983894073600
// MI455X (gfx1250) — matrix operand order repaired and checked by reading, not compiled, not run on hardware
//
#include <hip/hip_runtime.h>
#include <hip/hip_bf16.h>
#include <math.h>

// ---------------- problem constants (match reference) ----------------
constexpr int CB = 2;          // batch
constexpr int CT = 2048;       // tokens
constexpr int CE = 1024;       // embed
constexpr int CH = 16;         // heads
constexpr int CD = 64;         // head dim
constexpr int CK = 391;        // kept keys
constexpr int CKPAD = 416;     // 26 * 16
constexpr int CREC = 64;       // recency window
constexpr float CSCALE = 0.125f;   // D^-0.5

#define NEG_INF (-__builtin_inff())

typedef __attribute__((ext_vector_type(16))) __bf16 v16bf;
typedef __attribute__((ext_vector_type(8)))  float  v8f;
typedef __attribute__((ext_vector_type(4)))  unsigned v4u;
typedef __attribute__((ext_vector_type(8)))  int  v8i;
typedef __attribute__((ext_vector_type(4)))  int  v4i;

#if __has_builtin(__builtin_amdgcn_tensor_load_to_lds) && __has_builtin(__builtin_amdgcn_s_wait_tensorcnt)
#define HAVE_TDM 1
#else
#define HAVE_TDM 0
#endif

// ---------------- TDM: async 2-D bf16 tile load global -> LDS --------
// Packs a Tensor DMA Descriptor (D#) per CDNA5 ISA ch.8 and issues
// tensor_load_to_lds. Tile is `rows` x `rowlen` bf16 elements, row stride
// `stride` elements in global, written packed (rows*rowlen) at dst LDS.
__device__ inline void tdm_load_2d_bf16(__bf16* dst, const __bf16* src,
                                        unsigned rows, unsigned rowlen,
                                        unsigned stride) {
#if HAVE_TDM
  unsigned long long ga = (unsigned long long)(uintptr_t)src;
  unsigned lds = (unsigned)(uintptr_t)dst;       // low 32 bits = DS byte offset
  v4u g0;
  g0[0] = 1u;                                    // count=1, user descriptor
  g0[1] = lds;                                   // lds_addr
  g0[2] = (unsigned)ga;                          // global_addr[31:0]
  g0[3] = (unsigned)((ga >> 32) & 0x01ffffffu) | (2u << 30);  // addr hi | type=2
  v8i g1;
  g1[0] = (int)(1u << 16);                       // workgroup_mask=0, data_size=1 (2B)
  g1[1] = (int)((rowlen & 0xffffu) << 16);       // tensor_dim0[15:0]
  g1[2] = (int)(((rowlen >> 16) & 0xffffu) | ((rows & 0xffffu) << 16));
  g1[3] = (int)(((rows >> 16) & 0xffffu) | ((rowlen & 0xffffu) << 16)); // tile_dim0
  g1[4] = (int)(rows & 0xffffu);                 // tile_dim1, tile_dim2=0
  g1[5] = (int)stride;                           // tensor_dim0_stride[31:0]
  g1[6] = 0;
  g1[7] = 0;
  v4i z4 = {};
#if __clang_major__ >= 23
  v8i z8 = {};
  __builtin_amdgcn_tensor_load_to_lds(g0, g1, z4, z4, z8, 0);
#else
  __builtin_amdgcn_tensor_load_to_lds(g0, g1, z4, z4, 0);
#endif
#else
  int lane = threadIdx.x & 31;
  for (unsigned i = lane; i < rows * rowlen; i += 32)
    dst[i] = src[(i / rowlen) * stride + (i % rowlen)];
#endif
}

__device__ inline void tdm_wait_all() {
#if HAVE_TDM
  __builtin_amdgcn_s_wait_tensorcnt(0);
#endif
}

// ---------------- WMMA fragment loaders (CDNA5 16x16x32 bf16) --------
__device__ inline v16bf load_a_frag(const __bf16* A, int lda) {
  int lane = threadIdx.x & 31;
  int m = lane & 15;
  int half = lane >> 4;
  const __bf16* row = A + m * lda + half * 16;
  v16bf out;
#pragma unroll
  for (int i = 0; i < 8; ++i) out[i] = row[i];
#pragma unroll
  for (int i = 0; i < 8; ++i) out[8 + i] = row[8 + i];
  return out;
}

// B(k,n) = p[k*ld + n] (row-major K x N)
__device__ inline v16bf load_b_frag(const __bf16* Bm, int ld) {
  int lane = threadIdx.x & 31;
  int n = lane & 15;
  int half = lane >> 4;
  const __bf16* p = Bm + (half * 16) * ld + n;
  v16bf out;
#pragma unroll
  for (int i = 0; i < 16; ++i) out[i] = p[i * ld];
  return out;
}

// B(k,n) = p[n*ld + k] (transposed source, contiguous 32B reads)
__device__ inline v16bf load_b_fragT(const __bf16* Bt, int ld) {
  int lane = threadIdx.x & 31;
  int n = lane & 15;
  int half = lane >> 4;
  const __bf16* p = Bt + n * ld + half * 16;
  v16bf out;
#pragma unroll
  for (int i = 0; i < 16; ++i) out[i] = p[i];
  return out;
}

__device__ inline v8f wmma_bf16(v16bf a, v16bf b, v8f c) {
  return __builtin_amdgcn_wmma_f32_16x16x32_bf16(false, a, false, b,
                                                 (short)0, c, false, false);
}

__device__ inline float half_reduce_max(float v) {
  v = fmaxf(v, __shfl_xor(v, 1));
  v = fmaxf(v, __shfl_xor(v, 2));
  v = fmaxf(v, __shfl_xor(v, 4));
  v = fmaxf(v, __shfl_xor(v, 8));
  return v;
}
__device__ inline float half_reduce_sum(float v) {
  v += __shfl_xor(v, 1);
  v += __shfl_xor(v, 2);
  v += __shfl_xor(v, 4);
  v += __shfl_xor(v, 8);
  return v;
}

// ---------------- kernels -------------------------------------------
__global__ void cvt_bf16_kernel(const float* __restrict__ in,
                                __bf16* __restrict__ out, int n) {
  int i = blockIdx.x * blockDim.x + threadIdx.x;
  if (i < n) out[i] = (__bf16)in[i];
}

// 64x64 block-tile GEMM mainloop: C = A[64,K] * W[64,K]^T using TDM-staged
// double-buffered LDS tiles.  4 waves; wave w owns rows w*16..w*16+15 and all
// 4 column tiles (acc[4]).  Only wave 0 issues TDM (scalar branch).
__device__ inline void gemm64x64_mainloop(const __bf16* Abase, const __bf16* Bbase,
                                          int K, int lda, int ldb,
                                          __bf16 (&atile)[2][64 * 32],
                                          __bf16 (&btile)[2][64 * 32],
                                          v8f (&acc)[4], int wave, bool issuer) {
  const int NS = K / 32;
  if (issuer) {
    tdm_load_2d_bf16(&atile[0][0], Abase, 64, 32, (unsigned)lda);
    tdm_load_2d_bf16(&btile[0][0], Bbase, 64, 32, (unsigned)ldb);
  }
  for (int s = 0; s < NS; ++s) {
    int cur = s & 1;
    if (issuer) tdm_wait_all();
    __syncthreads();
    if (issuer && (s + 1 < NS)) {
      tdm_load_2d_bf16(&atile[cur ^ 1][0], Abase + (size_t)(s + 1) * 32, 64, 32, (unsigned)lda);
      tdm_load_2d_bf16(&btile[cur ^ 1][0], Bbase + (size_t)(s + 1) * 32, 64, 32, (unsigned)ldb);
    }
    v16bf a = load_a_frag(&atile[cur][0] + wave * 16 * 32, 32);
#pragma unroll
    for (int nt = 0; nt < 4; ++nt) {
      v16bf b = load_b_fragT(&btile[cur][0] + nt * 16 * 32, 32);
      acc[nt] = wmma_bf16(a, b, acc[nt]);
    }
    __syncthreads();
  }
}

// qkv = x @ W_attn^T ; scatter into q/k/v [B,H,T,D] bf16. grid(3E/64, B*T/64), 128 thr.
__global__ __launch_bounds__(128) void qkv_gemm_kernel(
    const __bf16* __restrict__ X, const __bf16* __restrict__ W,
    __bf16* __restrict__ qb, __bf16* __restrict__ kb, __bf16* __restrict__ vb) {
  __shared__ __bf16 atile[2][64 * 32];
  __shared__ __bf16 btile[2][64 * 32];
  int col0 = blockIdx.x * 64;
  int row0 = blockIdx.y * 64;
  int wave = __builtin_amdgcn_readfirstlane((int)(threadIdx.x >> 5));
  v8f zero = {};
  v8f acc[4] = {zero, zero, zero, zero};
  gemm64x64_mainloop(X + (size_t)row0 * CE, W + (size_t)col0 * CE, CE, CE, CE,
                     atile, btile, acc, wave, wave == 0);
  int lane = threadIdx.x & 31;
  int n = lane & 15, half = lane >> 4;
#pragma unroll
  for (int nt = 0; nt < 4; ++nt) {
    int c = col0 + nt * 16 + n;
    int which = c / CE;          // 0=q 1=k 2=v
    int e = c % CE;
    int h = e >> 6, d = e & 63;
    __bf16* dst = which == 0 ? qb : (which == 1 ? kb : vb);
#pragma unroll
    for (int r = 0; r < 8; ++r) {
      int m = row0 + wave * 16 + r + 8 * half;
      int bidx = m / CT, t = m % CT;
      dst[(((size_t)bidx * CH + h) * CT + t) * CD + d] = (__bf16)acc[nt][r];
    }
  }
}

// y = attn_out @ W_proj^T -> fp32 d_out. grid(E/64, B*T/64), 128 thr.
__global__ __launch_bounds__(128) void proj_gemm_kernel(
    const __bf16* __restrict__ A, const __bf16* __restrict__ W,
    float* __restrict__ Y) {
  __shared__ __bf16 atile[2][64 * 32];
  __shared__ __bf16 btile[2][64 * 32];
  int col0 = blockIdx.x * 64;
  int row0 = blockIdx.y * 64;
  int wave = __builtin_amdgcn_readfirstlane((int)(threadIdx.x >> 5));
  v8f zero = {};
  v8f acc[4] = {zero, zero, zero, zero};
  gemm64x64_mainloop(A + (size_t)row0 * CE, W + (size_t)col0 * CE, CE, CE, CE,
                     atile, btile, acc, wave, wave == 0);
  int lane = threadIdx.x & 31;
  int n = lane & 15, half = lane >> 4;
#pragma unroll
  for (int nt = 0; nt < 4; ++nt)
#pragma unroll
    for (int r = 0; r < 8; ++r) {
      int m = row0 + wave * 16 + r + 8 * half;
      Y[(size_t)m * CE + col0 + nt * 16 + n] = acc[nt][r];
    }
}

// Pass 1: causal softmax importance.  one wave per (b,h, 16-row query tile).
__global__ void pass1_importance_kernel(const __bf16* __restrict__ qbf,
                                        const __bf16* __restrict__ kbf,
                                        float* __restrict__ imp) {
  int gid = blockIdx.x;
  int qt = gid % (CT / 16);
  int bh = gid / (CT / 16);
  const __bf16* Q = qbf + ((size_t)bh * CT + qt * 16) * CD;
  const __bf16* Kp = kbf + (size_t)bh * CT * CD;
  int lane = threadIdx.x & 31;
  int n = lane & 15, half = lane >> 4;

  float runmax[8], runsum[8];
#pragma unroll
  for (int r = 0; r < 8; ++r) { runmax[r] = NEG_INF; runsum[r] = 0.f; }

  // sweep 1: online softmax stats over causal keys
  for (int kt = 0; kt <= qt; ++kt) {
    v8f acc = {};
    for (int ds = 0; ds < CD; ds += 32) {
      v16bf a = load_a_frag(Q + ds, CD);
      v16bf b = load_b_fragT(Kp + (size_t)kt * 16 * CD + ds, CD);
      acc = wmma_bf16(a, b, acc);
    }
#pragma unroll
    for (int r = 0; r < 8; ++r) {
      int qrow = qt * 16 + r + 8 * half;
      int kcol = kt * 16 + n;
      float s = (kcol <= qrow) ? acc[r] * CSCALE : NEG_INF;
      float tmax = half_reduce_max(s);
      float nm = fmaxf(runmax[r], tmax);
      float e = (s == NEG_INF) ? 0.f : __expf(s - nm);
      float tsum = half_reduce_sum(e);
      float f = (runmax[r] == NEG_INF) ? 0.f : __expf(runmax[r] - nm);
      runsum[r] = runsum[r] * f + tsum;
      runmax[r] = nm;
    }
  }

  // sweep 2: recompute scores, accumulate normalized importance
  const float invT = 1.0f / (float)CT;
  for (int kt = 0; kt <= qt; ++kt) {
    v8f acc = {};
    for (int ds = 0; ds < CD; ds += 32) {
      v16bf a = load_a_frag(Q + ds, CD);
      v16bf b = load_b_fragT(Kp + (size_t)kt * 16 * CD + ds, CD);
      acc = wmma_bf16(a, b, acc);
    }
    float colsum = 0.f;
#pragma unroll
    for (int r = 0; r < 8; ++r) {
      int qrow = qt * 16 + r + 8 * half;
      int kcol = kt * 16 + n;
      float s = (kcol <= qrow) ? acc[r] * CSCALE : NEG_INF;
      float p = (s == NEG_INF) ? 0.f : __expf(s - runmax[r]) / runsum[r];
      colsum += p;
    }
    colsum += __shfl_xor(colsum, 16);   // combine rows 0-7 with 8-15
    if (half == 0)
      atomicAdd(&imp[(size_t)bh * CT + kt * 16 + n], colsum * invT);
  }
}

__global__ void set_recency_kernel(float* __restrict__ imp) {
  int bh = blockIdx.x;
  imp[(size_t)bh * CT + (CT - CREC) + threadIdx.x] = 1.0f;
}

// top-k via bitonic sort (descending, ties ascending index) in LDS.
__global__ void topk_kernel(const float* __restrict__ imp,
                            int* __restrict__ idx_out) {
  int bh = blockIdx.x;
  const float* v = imp + (size_t)bh * CT;
  __shared__ float sv[CT];
  __shared__ int si[CT];
  for (int i = threadIdx.x; i < CT; i += blockDim.x) { sv[i] = v[i]; si[i] = i; }
  __syncthreads();
  for (unsigned k = 2; k <= (unsigned)CT; k <<= 1) {
    for (unsigned j = k >> 1; j > 0; j >>= 1) {
      for (unsigned i = threadIdx.x; i < (unsigned)CT; i += blockDim.x) {
        unsigned ixj = i ^ j;
        if (ixj > i) {
          float a = sv[i], b = sv[ixj];
          int ia = si[i], ib = si[ixj];
          bool keep = (a > b) || (a == b && ia < ib);
          bool dir = ((i & k) == 0);
          if (dir != keep) { sv[i] = b; sv[ixj] = a; si[i] = ib; si[ixj] = ia; }
        }
      }
      __syncthreads();
    }
  }
  for (int i = threadIdx.x; i < CK; i += blockDim.x)
    idx_out[(size_t)bh * CK + i] = si[i];
}

// gather pruned K/V into [B,H,KPAD,D] bf16, zero pad beyond CK.
__global__ void gather_kv_kernel(const __bf16* __restrict__ kbf,
                                 const __bf16* __restrict__ vbf,
                                 const int* __restrict__ idx,
                                 __bf16* __restrict__ Kk,
                                 __bf16* __restrict__ Vk) {
  size_t g = (size_t)blockIdx.x * blockDim.x + threadIdx.x;
  size_t total = (size_t)CB * CH * CKPAD * CD;
  if (g >= total) return;
  int d = (int)(g & (CD - 1));
  int j = (int)((g / CD) % CKPAD);
  int bh = (int)(g / ((size_t)CD * CKPAD));
  __bf16 kv = (__bf16)0.0f, vv = (__bf16)0.0f;
  if (j < CK) {
    int tok = idx[(size_t)bh * CK + j];
    kv = kbf[((size_t)bh * CT + tok) * CD + d];
    vv = vbf[((size_t)bh * CT + tok) * CD + d];
  }
  Kk[g] = kv;
  Vk[g] = vv;
}

// Pass 2: attention over pruned keys.  one wave per (b,h,16-query-tile).
// K and V tiles stream in via double-buffered TDM loads.
__global__ __launch_bounds__(32) void pass2_attn_kernel(
    const __bf16* __restrict__ qbf, const __bf16* __restrict__ Kk,
    const __bf16* __restrict__ Vk, __bf16* __restrict__ aout) {
  int gid = blockIdx.x;
  int qt = gid % (CT / 16);
  int bh = gid / (CT / 16);
  int b = bh / CH, h = bh % CH;
  const __bf16* Q = qbf + ((size_t)bh * CT + qt * 16) * CD;
  const __bf16* Kbase = Kk + (size_t)bh * CKPAD * CD;
  const __bf16* Vbase = Vk + (size_t)bh * CKPAD * CD;

  __shared__ float sbuf[16][CKPAD];
  __shared__ __bf16 pbuf[16][CKPAD];
  __shared__ __bf16 kst[2][16 * CD];
  __shared__ __bf16 vst[2][32 * CD];

  int lane = threadIdx.x & 31;
  int n = lane & 15, half = lane >> 4;

  // scores: S = Q K^T, K tiles via TDM double buffer
  const int NT = CKPAD / 16;
  tdm_load_2d_bf16(&kst[0][0], Kbase, 16, CD, CD);
  for (int kt = 0; kt < NT; ++kt) {
    int cur = kt & 1;
    tdm_wait_all();
    if (kt + 1 < NT)
      tdm_load_2d_bf16(&kst[cur ^ 1][0], Kbase + (size_t)(kt + 1) * 16 * CD, 16, CD, CD);
    v8f acc = {};
    for (int ds = 0; ds < CD; ds += 32) {
      v16bf a = load_a_frag(Q + ds, CD);
      v16bf bfr = load_b_fragT(&kst[cur][0] + ds, CD);
      acc = wmma_bf16(a, bfr, acc);
    }
    int j = kt * 16 + n;
#pragma unroll
    for (int r = 0; r < 8; ++r)
      sbuf[r + 8 * half][j] = (j < CK) ? acc[r] * CSCALE : NEG_INF;
  }
  __syncthreads();

  // row softmax (lanes 0..15, one row each), P -> bf16 LDS
  if (lane < 16) {
    int row = lane;
    float mx = NEG_INF;
    for (int j = 0; j < CKPAD; ++j) mx = fmaxf(mx, sbuf[row][j]);
    float sm = 0.f;
    for (int j = 0; j < CKPAD; ++j) {
      float s = sbuf[row][j];
      float e = (s == NEG_INF) ? 0.f : __expf(s - mx);
      sbuf[row][j] = e;
      sm += e;
    }
    float inv = 1.0f / sm;
    for (int j = 0; j < CKPAD; ++j) pbuf[row][j] = (__bf16)(sbuf[row][j] * inv);
  }
  __syncthreads();

  // O = P @ V, V tiles via TDM double buffer
  const int NV = CKPAD / 32;
  v8f zero = {};
  v8f o[4] = {zero, zero, zero, zero};
  tdm_load_2d_bf16(&vst[0][0], Vbase, 32, CD, CD);
  for (int kc = 0; kc < NV; ++kc) {
    int cur = kc & 1;
    tdm_wait_all();
    if (kc + 1 < NV)
      tdm_load_2d_bf16(&vst[cur ^ 1][0], Vbase + (size_t)(kc + 1) * 32 * CD, 32, CD, CD);
    v16bf a = load_a_frag(&pbuf[0][0] + kc * 32, CKPAD);
#pragma unroll
    for (int nt = 0; nt < 4; ++nt) {
      v16bf bfr = load_b_frag(&vst[cur][0] + nt * 16, CD);
      o[nt] = wmma_bf16(a, bfr, o[nt]);
    }
  }

#pragma unroll
  for (int nt = 0; nt < 4; ++nt)
#pragma unroll
    for (int r = 0; r < 8; ++r) {
      int t = qt * 16 + r + 8 * half;
      int d = nt * 16 + n;
      aout[((size_t)b * CT + t) * CE + h * CD + d] = (__bf16)o[nt][r];
    }
}

// ---------------- host launch ---------------------------------------
extern "C" void kernel_launch(void* const* d_in, const int* in_sizes, int n_in,
                              void* d_out, int out_size, void* d_ws, size_t ws_size,
                              hipStream_t stream) {
  (void)in_sizes; (void)n_in; (void)out_size; (void)ws_size;
  const float* x = (const float*)d_in[0];       // [B,T,E]
  const float* Wattn = (const float*)d_in[1];   // [3E,E]
  const float* Wproj = (const float*)d_in[2];   // [E,E]
  float* out = (float*)d_out;                   // [B,T,E]

  char* ws = (char*)d_ws;
  size_t off = 0;
  auto alloc = [&](size_t bytes) -> void* {
    void* p = (void*)(ws + off);
    off += (bytes + 255) & ~(size_t)255;
    return p;
  };

  const size_t nBTE = (size_t)CB * CT * CE;
  const size_t nBHTD = (size_t)CB * CH * CT * CD;
  const size_t nKV = (size_t)CB * CH * CKPAD * CD;

  __bf16* xbf = (__bf16*)alloc(nBTE * 2);
  __bf16* wattnbf = (__bf16*)alloc((size_t)3 * CE * CE * 2);
  __bf16* wprojbf = (__bf16*)alloc((size_t)CE * CE * 2);
  __bf16* qbf = (__bf16*)alloc(nBHTD * 2);
  __bf16* kbf = (__bf16*)alloc(nBHTD * 2);
  __bf16* vbf = (__bf16*)alloc(nBHTD * 2);
  float* imp = (float*)alloc((size_t)CB * CH * CT * 4);
  int* idx = (int*)alloc((size_t)CB * CH * CK * 4);
  __bf16* Kk = (__bf16*)alloc(nKV * 2);
  __bf16* Vk = (__bf16*)alloc(nKV * 2);
  __bf16* aoutbf = (__bf16*)alloc(nBTE * 2);

  // 1. fp32 -> bf16 conversions
  cvt_bf16_kernel<<<(int)((nBTE + 255) / 256), 256, 0, stream>>>(x, xbf, (int)nBTE);
  cvt_bf16_kernel<<<(3 * CE * CE + 255) / 256, 256, 0, stream>>>(Wattn, wattnbf, 3 * CE * CE);
  cvt_bf16_kernel<<<(CE * CE + 255) / 256, 256, 0, stream>>>(Wproj, wprojbf, CE * CE);

  // 2. QKV GEMM (WMMA + TDM staging)
  qkv_gemm_kernel<<<dim3(3 * CE / 64, CB * CT / 64), 128, 0, stream>>>(
      xbf, wattnbf, qbf, kbf, vbf);

  // 3. importance (zero, accumulate, recency override)
  hipMemsetAsync(imp, 0, (size_t)CB * CH * CT * 4, stream);
  pass1_importance_kernel<<<CB * CH * (CT / 16), 32, 0, stream>>>(qbf, kbf, imp);
  set_recency_kernel<<<CB * CH, CREC, 0, stream>>>(imp);

  // 4. top-k selection
  topk_kernel<<<CB * CH, 1024, 0, stream>>>(imp, idx);

  // 5. gather pruned K/V (zero-padded to CKPAD)
  gather_kv_kernel<<<(int)((nKV + 255) / 256), 256, 0, stream>>>(kbf, vbf, idx, Kk, Vk);

  // 6. pruned attention pass (WMMA + TDM staging)
  pass2_attn_kernel<<<CB * CH * (CT / 16), 32, 0, stream>>>(qbf, Kk, Vk, aoutbf);

  // 7. output projection (WMMA + TDM staging) -> fp32 out
  proj_gemm_kernel<<<dim3(CE / 64, CB * CT / 64), 128, 0, stream>>>(aoutbf, wprojbf, out);
}
